// Relationnet_44074954391641
// MI455X (gfx1250) — matrix operand order repaired and checked by reading, not compiled, not run on hardware
//
#include <hip/hip_runtime.h>

#define F_DIM 1024
#define N_DIM 2048
#define B_DIM 8
#define KT 32
#define BM 128
#define BN 128
#define LDT 34   // padded LDS row stride in u16 (34*2=68B, 4B aligned, 17 dwords -> bank-friendly)

typedef __attribute__((ext_vector_type(16))) __bf16 v16bf;
typedef __attribute__((ext_vector_type(8)))  float  v8f;

union Frag { v16bf v; unsigned int u[8]; };

__device__ __forceinline__ void split_bf16(float x, unsigned short& hi, unsigned short& lo) {
    unsigned int ux = __float_as_uint(x);
    unsigned short h = (unsigned short)(ux >> 16);          // truncate to bf16
    float hf = __uint_as_float(((unsigned int)h) << 16);
    float r  = x - hf;                                      // residual, |r| < 2^-7 |x|
    hi = h;
    lo = (unsigned short)(__float_as_uint(r) >> 16);
}

// ---------------- norms: nrm[b, c] = ||X[b, :, c]||_2 over f ----------------
__global__ void k_norms(const float* __restrict__ prev, const float* __restrict__ curr,
                        float* __restrict__ nrm1, float* __restrict__ nrm2) {
    int gid  = blockIdx.x * blockDim.x + threadIdx.x;   // 0..32767
    int half = gid >> 14;                               // 0: prev, 1: curr
    int idx  = gid & 16383;
    int b    = idx >> 11;
    int c    = idx & 2047;
    const float* src = half ? curr : prev;
    const float* col = src + (size_t)b * F_DIM * N_DIM + c;
    float s = 0.f;
    for (int k = 0; k < F_DIM; ++k) { float v = col[(size_t)k * N_DIM]; s += v * v; }
    float n = sqrtf(s);
    if (half) nrm2[idx] = n; else nrm1[idx] = n;
}

// ---------------- WMMA fragment loads from LDS ----------------
// 16-bit A-matrix 16x32 layout: lane L holds row M=L&15; lanes 0-15: VGPR0-3 -> K=0..7,
// VGPR4-7 -> K=16..23; lanes 16-31: +8. Each VGPR packs (K, K+1) as low/high bf16.
__device__ __forceinline__ void load_frag_a(const unsigned short* base, int row0, int lane, Frag& f) {
    const unsigned short* row = base + (row0 + (lane & 15)) * LDT;
    int hoff = (lane >> 4) << 3;  // 0 or 8
#pragma unroll
    for (int r = 0; r < 8; ++r) {
        int k = ((r < 4) ? (r << 1) : (16 + ((r - 4) << 1))) + hoff;
        f.u[r] = *(const unsigned int*)(row + k);
    }
}
// 16-bit B-matrix 32x16: lane L holds col N=L&15; lanes 0-15: VGPR0-3 -> K=0..7,
// VGPR4-7 -> K=16..23; lanes 16-31: +8.
__device__ __forceinline__ void load_frag_b(const unsigned short* base, int row0, int lane, Frag& f) {
    const unsigned short* row = base + (row0 + (lane & 15)) * LDT;
    int hoff = (lane >> 4) << 3;  // 0 or 8
#pragma unroll
    for (int r = 0; r < 8; ++r) {
        int k = ((r < 4) ? (r << 1) : (16 + ((r - 4) << 1))) + hoff;
        f.u[r] = *(const unsigned int*)(row + k);
    }
}

// ---------------- fused similarity GEMM + per-block argmax ----------------
// Block tile 128x128, 8 waves in 4(M) x 2(N); wave tile 32x64 = 2 x 4 WMMA frags.
__global__ __launch_bounds__(256)
void k_sim_argmax(const float* __restrict__ prev, const float* __restrict__ curr,
                  const float* __restrict__ nrm1, const float* __restrict__ nrm2,
                  float4* __restrict__ partials) {
    __shared__ unsigned short Ahi[BM * LDT];
    __shared__ unsigned short Alo[BM * LDT];
    __shared__ unsigned short Bh_[BN * LDT];
    __shared__ unsigned short Bl_[BN * LDT];
    __shared__ float redV[256];
    __shared__ int   redM[256];
    __shared__ int   redN[256];

    const int tid  = threadIdx.x;
    const int lane = tid & 31;
    const int wave = tid >> 5;      // 0..7
    const int wm   = wave >> 1;     // 0..3  (M direction, 32 rows each)
    const int wn   = wave & 1;      // 0..1  (N direction, 64 cols each)
    const int b    = blockIdx.z;
    const int m0   = blockIdx.y * BM;
    const int n0   = blockIdx.x * BN;

    const float* Ag = prev + (size_t)b * F_DIM * N_DIM + m0;   // element (k,m): Ag[k*N_DIM + m]
    const float* Bg = curr + (size_t)b * F_DIM * N_DIM + n0;

    v8f acc[2][4];
#pragma unroll
    for (int mf = 0; mf < 2; ++mf)
#pragma unroll
        for (int nf = 0; nf < 4; ++nf)
#pragma unroll
            for (int e = 0; e < 8; ++e) acc[mf][nf][e] = 0.f;

    for (int k0 = 0; k0 < F_DIM; k0 += KT) {
        // ---- stage A tile (32x128 f32 -> hi/lo bf16 pairs), 2 groups/thread ----
#pragma unroll
        for (int g = 0; g < 2; ++g) {
            int grp = tid + (g << 8);          // 0..511
            int k   = (grp >> 5) << 1;         // even k
            int m4  = (grp & 31) << 2;
            const float* p0 = Ag + (size_t)(k0 + k) * N_DIM + m4;
            float4 a0 = *(const float4*)p0;
            float4 a1 = *(const float4*)(p0 + N_DIM);
#pragma unroll
            for (int j = 0; j < 4; ++j) {
                float x0 = ((const float*)&a0)[j];
                float x1 = ((const float*)&a1)[j];
                unsigned short h0, l0, h1, l1;
                split_bf16(x0, h0, l0); split_bf16(x1, h1, l1);
                int off = (m4 + j) * LDT + k;
                *(unsigned int*)(Ahi + off) = (unsigned int)h0 | ((unsigned int)h1 << 16);
                *(unsigned int*)(Alo + off) = (unsigned int)l0 | ((unsigned int)l1 << 16);
            }
        }
        // ---- stage B tile (32x128), 2 groups/thread ----
#pragma unroll
        for (int g = 0; g < 2; ++g) {
            int grp = tid + (g << 8);
            int k   = (grp >> 5) << 1;
            int n4  = (grp & 31) << 2;
            const float* p0 = Bg + (size_t)(k0 + k) * N_DIM + n4;
            float4 c0 = *(const float4*)p0;
            float4 c1 = *(const float4*)(p0 + N_DIM);
#pragma unroll
            for (int j = 0; j < 4; ++j) {
                float x0 = ((const float*)&c0)[j];
                float x1 = ((const float*)&c1)[j];
                unsigned short h0, l0, h1, l1;
                split_bf16(x0, h0, l0); split_bf16(x1, h1, l1);
                int off = (n4 + j) * LDT + k;
                *(unsigned int*)(Bh_ + off) = (unsigned int)h0 | ((unsigned int)h1 << 16);
                *(unsigned int*)(Bl_ + off) = (unsigned int)l0 | ((unsigned int)l1 << 16);
            }
        }
        // prefetch next K tile while this one computes
        if (k0 + KT < F_DIM) {
            __builtin_prefetch(Ag + (size_t)(k0 + KT + (tid >> 3)) * N_DIM + (tid & 7) * 16, 0, 0);
            __builtin_prefetch(Bg + (size_t)(k0 + KT + (tid >> 3)) * N_DIM + (tid & 7) * 16, 0, 0);
        }
        __syncthreads();

        Frag ah[2], al[2], bh[4], bl[4];
#pragma unroll
        for (int mf = 0; mf < 2; ++mf) {
            load_frag_a(Ahi, wm * 32 + mf * 16, lane, ah[mf]);
            load_frag_a(Alo, wm * 32 + mf * 16, lane, al[mf]);
        }
#pragma unroll
        for (int nf = 0; nf < 4; ++nf) {
            load_frag_b(Bh_, wn * 64 + nf * 16, lane, bh[nf]);
            load_frag_b(Bl_, wn * 64 + nf * 16, lane, bl[nf]);
        }
#pragma unroll
        for (int mf = 0; mf < 2; ++mf)
#pragma unroll
            for (int nf = 0; nf < 4; ++nf) {
                // near-fp32 product: hi*lo + lo*hi + hi*hi
                acc[mf][nf] = __builtin_amdgcn_wmma_f32_16x16x32_bf16(
                    false, ah[mf].v, false, bl[nf].v, (short)0, acc[mf][nf], false, false);
                acc[mf][nf] = __builtin_amdgcn_wmma_f32_16x16x32_bf16(
                    false, al[mf].v, false, bh[nf].v, (short)0, acc[mf][nf], false, false);
                acc[mf][nf] = __builtin_amdgcn_wmma_f32_16x16x32_bf16(
                    false, ah[mf].v, false, bh[nf].v, (short)0, acc[mf][nf], false, false);
            }
        __syncthreads();
    }

    // ---- epilogue: normalize + argmax (C/D layout: M = e + 8*(lane>>4), N = lane&15) ----
    float bestv = -3.4e38f; int bestm = 0, bestn = 0;
    const int half8 = (lane >> 4) << 3;
#pragma unroll
    for (int mf = 0; mf < 2; ++mf)
#pragma unroll
        for (int nf = 0; nf < 4; ++nf)
#pragma unroll
            for (int e = 0; e < 8; ++e) {
                int gm = m0 + wm * 32 + mf * 16 + e + half8;
                int gn = n0 + wn * 64 + nf * 16 + (lane & 15);
                float den = fmaxf(nrm1[b * N_DIM + gm] * nrm2[b * N_DIM + gn], 1e-8f);
                float s = acc[mf][nf][e] / den;
                if (s > bestv || (s == bestv && (gm < bestm || (gm == bestm && gn < bestn)))) {
                    bestv = s; bestm = gm; bestn = gn;
                }
            }
    redV[tid] = bestv; redM[tid] = bestm; redN[tid] = bestn;
    __syncthreads();
    for (int off = 128; off > 0; off >>= 1) {
        if (tid < off) {
            float v2 = redV[tid + off]; int m2 = redM[tid + off], q2 = redN[tid + off];
            float v1 = redV[tid];       int m1 = redM[tid],       q1 = redN[tid];
            if (v2 > v1 || (v2 == v1 && (m2 < m1 || (m2 == m1 && q2 < q1)))) {
                redV[tid] = v2; redM[tid] = m2; redN[tid] = q2;
            }
        }
        __syncthreads();
    }
    if (tid == 0) {
        int idx = (b * gridDim.y + blockIdx.y) * gridDim.x + blockIdx.x;
        partials[idx] = make_float4(redV[0], (float)redM[0], (float)redN[0], 0.f);
    }
}

// ---------------- per-batch argmax reduce + feature gather ----------------
__global__ void k_reduce_gather(const float4* __restrict__ partials,
                                const float* __restrict__ prev, const float* __restrict__ curr,
                                float* __restrict__ fusion) {
    __shared__ float redV[256];
    __shared__ int   redM[256];
    __shared__ int   redN[256];
    __shared__ int   selM, selN;
    const int b = blockIdx.x, tid = threadIdx.x;
    const int per = (N_DIM / BM) * (N_DIM / BN);   // 256 tiles per batch
    float bv = -3.4e38f; int bm = 0, bn = 0;
    for (int i = tid; i < per; i += 256) {
        float4 p = partials[b * per + i];
        int m = (int)p.y, n = (int)p.z;
        if (p.x > bv || (p.x == bv && (m < bm || (m == bm && n < bn)))) { bv = p.x; bm = m; bn = n; }
    }
    redV[tid] = bv; redM[tid] = bm; redN[tid] = bn;
    __syncthreads();
    for (int off = 128; off > 0; off >>= 1) {
        if (tid < off) {
            float v2 = redV[tid + off]; int m2 = redM[tid + off], q2 = redN[tid + off];
            float v1 = redV[tid];       int m1 = redM[tid],       q1 = redN[tid];
            if (v2 > v1 || (v2 == v1 && (m2 < m1 || (m2 == m1 && q2 < q1)))) {
                redV[tid] = v2; redM[tid] = m2; redN[tid] = q2;
            }
        }
        __syncthreads();
    }
    if (tid == 0) { selM = redM[0]; selN = redN[0]; }
    __syncthreads();
    for (int t = tid; t < 2 * F_DIM; t += 256) {
        float v;
        if (t < F_DIM) v = prev[(size_t)b * F_DIM * N_DIM + (size_t)t * N_DIM + selM];
        else           v = curr[(size_t)b * F_DIM * N_DIM + (size_t)(t - F_DIM) * N_DIM + selN];
        fusion[b * (2 * F_DIM) + t] = v;
    }
}

// ---------------- MLP: wave-per-batch dot products ----------------
__global__ void k_lin(const float* __restrict__ X, const float* __restrict__ W,
                      const float* __restrict__ bias, float* __restrict__ Y,
                      int in_dim, int out_dim) {
    int j = blockIdx.x;                 // output neuron
    int w = threadIdx.x >> 5;           // batch 0..7 (8 waves)
    int l = threadIdx.x & 31;
    const float* x  = X + (size_t)w * in_dim;
    const float* wr = W + (size_t)j * in_dim;
    float s = 0.f;
    for (int i = l; i < in_dim; i += 32) s += x[i] * wr[i];
    for (int off = 16; off > 0; off >>= 1) s += __shfl_xor(s, off, 32);
    if (l == 0) Y[w * out_dim + j] = s + bias[j];
}

__global__ void k_bn_relu(float* __restrict__ h, const float* __restrict__ g,
                          const float* __restrict__ bt, int n_feat) {
    int j = threadIdx.x;
    if (j >= n_feat) return;
    float v[8]; float mu = 0.f;
    for (int bb = 0; bb < 8; ++bb) { v[bb] = h[bb * n_feat + j]; mu += v[bb]; }
    mu *= 0.125f;
    float var = 0.f;
    for (int bb = 0; bb < 8; ++bb) { float d = v[bb] - mu; var += d * d; }
    var *= 0.125f;                                   // biased variance (torch BN forward)
    float sc = rsqrtf(var + 1e-5f) * g[j];
    float sh = bt[j];
    for (int bb = 0; bb < 8; ++bb) {
        float t = (v[bb] - mu) * sc + sh;
        h[bb * n_feat + j] = fmaxf(t, 0.f);
    }
}

__global__ void k_fc(const float* __restrict__ A2, const float* __restrict__ Wfc,
                     const float* __restrict__ bfc, float* __restrict__ out) {
    int t = threadIdx.x;
    if (t >= 16) return;
    int bb = t >> 1, o = t & 1;
    const float* x  = A2 + bb * 256;
    const float* wr = Wfc + o * 256;
    float s = 0.f;
    for (int i = 0; i < 256; ++i) s += x[i] * wr[i];
    out[bb * 2 + o] = s + bfc[o];
}

extern "C" void kernel_launch(void* const* d_in, const int* in_sizes, int n_in,
                              void* d_out, int out_size, void* d_ws, size_t ws_size,
                              hipStream_t stream) {
    (void)in_sizes; (void)n_in; (void)out_size; (void)ws_size;
    const float* prev = (const float*)d_in[0];
    const float* curr = (const float*)d_in[1];
    const float* W1   = (const float*)d_in[2];
    const float* b1   = (const float*)d_in[3];
    const float* g1   = (const float*)d_in[4];
    const float* bt1  = (const float*)d_in[5];
    const float* W2   = (const float*)d_in[6];
    const float* b2   = (const float*)d_in[7];
    const float* g2   = (const float*)d_in[8];
    const float* bt2  = (const float*)d_in[9];
    const float* Wfc  = (const float*)d_in[10];
    const float* bfc  = (const float*)d_in[11];
    float* out = (float*)d_out;

    float*  ws       = (float*)d_ws;
    float*  nrm1     = ws;                       // 16384
    float*  nrm2     = ws + 16384;               // 16384
    float4* partials = (float4*)(ws + 32768);    // 2048 float4 = 8192 floats
    float*  fusion   = ws + 40960;               // 16384
    float*  h1       = ws + 57344;               // 4096
    float*  h2       = ws + 61440;               // 2048

    k_norms<<<128, 256, 0, stream>>>(prev, curr, nrm1, nrm2);

    dim3 g(N_DIM / BN, N_DIM / BM, B_DIM);       // (16, 16, 8)
    k_sim_argmax<<<g, 256, 0, stream>>>(prev, curr, nrm1, nrm2, partials);

    k_reduce_gather<<<8, 256, 0, stream>>>(partials, prev, curr, fusion);

    k_lin<<<512, 256, 0, stream>>>(fusion, W1, b1, h1, 2048, 512);
    k_bn_relu<<<1, 512, 0, stream>>>(h1, g1, bt1, 512);
    k_lin<<<256, 256, 0, stream>>>(h1, W2, b2, h2, 512, 256);
    k_bn_relu<<<1, 256, 0, stream>>>(h2, g2, bt2, 256);
    k_fc<<<1, 64, 0, stream>>>(h2, Wfc, bfc, out);
}
